// CapsuleNetwork_90391881712310
// MI455X (gfx1250) — matrix operand order repaired and checked by reading, not compiled, not run on hardware
//
#include <hip/hip_runtime.h>
#include <hip/hip_bf16.h>
#include <math.h>
#include <stdint.h>

// ---------- CDNA5 WMMA types ----------
typedef __attribute__((ext_vector_type(16))) __bf16 v16bf;
typedef __attribute__((ext_vector_type(8)))  float  v8f;

struct Pack32 { uint4 lo; uint4 hi; };   // 32 bytes == one v16bf fragment

__device__ __forceinline__ unsigned short f2bf(float f) {
    unsigned int u = __float_as_uint(f);
    u = (u + 0x7FFFu + ((u >> 16) & 1u)) >> 16;   // RNE truncate to bf16
    return (unsigned short)u;
}

// LDS byte offset of a __shared__ object = low 32 bits of its generic address
// (shared aperture encodes the LDS offset in addr[31:0]).
__device__ __forceinline__ unsigned ldsOffset(const void* p) {
    return (unsigned)(unsigned long long)(uintptr_t)p;
}

// gfx1250 async DMA: global -> LDS, 16 bytes per lane, tracked by ASYNCcnt.
__device__ __forceinline__ void asyncLoadB128(unsigned ldsOff, const void* g) {
    asm volatile("global_load_async_to_lds_b128 %0, %1, off"
                 :: "v"(ldsOff), "v"((unsigned long long)(uintptr_t)g)
                 : "memory");
}

// ================= layout conversion =================
// x: [32,3,64,64] f32  ->  xb: [32,64,64,3] bf16
__global__ void cvt_x_k(const float* __restrict__ x, unsigned short* __restrict__ xb, int total) {
    int i = blockIdx.x * blockDim.x + threadIdx.x;
    if (i >= total) return;
    int c = i % 3, xx = (i / 3) & 63, yy = (i / 192) & 63, b = i / 12288;
    xb[i] = f2bf(x[((b * 3 + c) * 64 + yy) * 64 + xx]);
}

// conv1_w: [256,3,9,9] f32 -> w1t: [256][256] bf16, k=(ky*9+kx)*3+c, padded 243->256 with 0
__global__ void cvt_w1_k(const float* __restrict__ w, unsigned short* __restrict__ wt) {
    int i = blockIdx.x * blockDim.x + threadIdx.x;
    if (i >= 256 * 256) return;
    int o = i >> 8, k = i & 255;
    unsigned short v = 0;
    if (k < 243) {
        int ky = k / 27, t = k % 27, kx = t / 3, c = t % 3;
        v = f2bf(w[((o * 3 + c) * 9 + ky) * 9 + kx]);
    }
    wt[i] = v;
}

// conv2_w: [256,256,9,9] f32 -> w2t: [256][20736] bf16, k=(ky*9+kx)*256+c
__global__ void cvt_w2_k(const float* __restrict__ w, unsigned short* __restrict__ wt, int total) {
    int i = blockIdx.x * blockDim.x + threadIdx.x;
    if (i >= total) return;
    int o = i / 20736, k = i % 20736;
    int ky = k / 2304, kx = (k / 256) % 9, c = k & 255;
    wt[i] = f2bf(w[((o * 256 + c) * 9 + ky) * 9 + kx]);
}

__global__ void zero_f32_k(float* __restrict__ p, int total) {
    int i = blockIdx.x * blockDim.x + threadIdx.x;
    if (i < total) p[i] = 0.f;
}

// ================= WMMA implicit-GEMM conv1 =================
// A: im2col of xb[32,64,64,3], M=100352, K=256(pad 243). B: w1t[N=256][K=256].
// Out: h1[m][n] bf16 NHWC ([b][56][56][256]), relu(acc+bias).
__global__ __launch_bounds__(256) void conv1_wmma_k(
    const unsigned short* __restrict__ xb,
    const unsigned short* __restrict__ w1t,
    const float* __restrict__ bias,
    unsigned short* __restrict__ h1)
{
    __shared__ unsigned short As[128 * 40];
    __shared__ unsigned short Bs[128 * 40];
    const int m0 = blockIdx.x * 128;
    const int n0 = blockIdx.y * 128;
    const int tid = threadIdx.x;
    const int lane = tid & 31;
    const int wid  = tid >> 5;
    const int wm = wid & 3;    // 4 waves over M (32 rows each)
    const int wn = wid >> 2;   // 2 waves over N (64 cols each)

    v8f acc[2][4];
    v8f z = {0.f, 0.f, 0.f, 0.f, 0.f, 0.f, 0.f, 0.f};
    for (int i = 0; i < 2; ++i) for (int j = 0; j < 4; ++j) acc[i][j] = z;

    const int row  = tid & 127;
    const int half = tid >> 7;
    const int m  = m0 + row;
    const int b  = m / 3136;
    const int rm = m % 3136;
    const int oy = rm / 56;
    const int ox = rm % 56;
    const unsigned short* wrow = w1t + (size_t)(n0 + row) * 256 + half * 16;

    for (int k0 = 0; k0 < 256; k0 += 32) {
        // ---- A tile: scalar im2col gather (K order (ky*9+kx)*3+c, guard k<243)
        #pragma unroll
        for (int j = 0; j < 16; ++j) {
            int k = k0 + half * 16 + j;
            unsigned short v = 0;
            if (k < 243) {
                int ky = k / 27, t = k % 27, kx = t / 3, c = t % 3;
                v = xb[((b * 64 + oy + ky) * 64 + (ox + kx)) * 3 + c];
            }
            As[row * 40 + half * 16 + j] = v;
        }
        // ---- B tile: vectorized (16B) loads, K contiguous per n row
        {
            const uint4* src = (const uint4*)(wrow + k0);
            *(uint4*)&Bs[row * 40 + half * 16 + 0] = src[0];
            *(uint4*)&Bs[row * 40 + half * 16 + 8] = src[1];
        }
        __syncthreads();

        v16bf aF[2], bF[4];
        #pragma unroll
        for (int i = 0; i < 2; ++i) {  // A 16x32 bf16 fragment: 16 contiguous K per lane
            int ml = wm * 32 + i * 16 + (lane & 15);
            int kh = lane >> 4;
            Pack32 p;
            p.lo = *(const uint4*)&As[ml * 40 + kh * 16];
            p.hi = *(const uint4*)&As[ml * 40 + kh * 16 + 8];
            aF[i] = __builtin_bit_cast(v16bf, p);
        }
        #pragma unroll
        for (int j = 0; j < 4; ++j) {  // B 32x16 bf16 fragment: 16 contiguous K per lane
            int nl = wn * 64 + j * 16 + (lane & 15);
            int ks = (lane >> 4) * 16;
            Pack32 p;
            p.lo = *(const uint4*)&Bs[nl * 40 + ks];
            p.hi = *(const uint4*)&Bs[nl * 40 + ks + 8];
            bF[j] = __builtin_bit_cast(v16bf, p);
        }
        #pragma unroll
        for (int i = 0; i < 2; ++i)
            #pragma unroll
            for (int j = 0; j < 4; ++j)
                acc[i][j] = __builtin_amdgcn_wmma_f32_16x16x32_bf16(
                    false, aF[i], false, bF[j], (short)0, acc[i][j], false, false);
        __syncthreads();
    }

    // epilogue: bias + relu -> bf16 NHWC
    #pragma unroll
    for (int i = 0; i < 2; ++i)
        #pragma unroll
        for (int j = 0; j < 4; ++j)
            #pragma unroll
            for (int v = 0; v < 8; ++v) {
                int mg = m0 + wm * 32 + i * 16 + v + (lane >> 4) * 8;
                int ng = n0 + wn * 64 + j * 16 + (lane & 15);
                float r = acc[i][j][v] + bias[ng];
                r = r > 0.f ? r : 0.f;
                h1[(size_t)mg * 256 + ng] = f2bf(r);
            }
}

// ================= WMMA implicit-GEMM conv2 (stride 2) =================
// A: im2col of h1[32,56,56,256] bf16, M=18432, K=20736 (k=(ky*9+kx)*256+c).
// B: w2t[N=256][K=20736]. Out: h2[m][n] f32 NHWC ([b][24][24][256]) = acc+bias.
// Double-buffered LDS staging via gfx1250 global_load_async_to_lds_b128 (ASYNCcnt).
__global__ __launch_bounds__(256) void conv2_wmma_k(
    const unsigned short* __restrict__ h1,
    const unsigned short* __restrict__ w2t,
    const float* __restrict__ bias,
    float* __restrict__ h2)
{
    __shared__ unsigned short As[2][128 * 40];
    __shared__ unsigned short Bs[2][128 * 40];
    const int m0 = blockIdx.x * 128;
    const int n0 = blockIdx.y * 128;
    const int tid = threadIdx.x;
    const int lane = tid & 31;
    const int wid  = tid >> 5;
    const int wm = wid & 3;
    const int wn = wid >> 2;

    v8f acc[2][4];
    v8f z = {0.f, 0.f, 0.f, 0.f, 0.f, 0.f, 0.f, 0.f};
    for (int i = 0; i < 2; ++i) for (int j = 0; j < 4; ++j) acc[i][j] = z;

    const int row  = tid & 127;
    const int half = tid >> 7;
    const int m  = m0 + row;
    const int b  = m / 576;
    const int rm = m % 576;
    const int oy = rm / 24;
    const int ox = rm % 24;
    const unsigned short* wrow = w2t + (size_t)(n0 + row) * 20736 + half * 16;

    // per-thread LDS destination byte-offsets for the DMA (2 x 16B for A, 2 x 16B for B)
    unsigned aOff[2], bOff[2];
    #pragma unroll
    for (int u = 0; u < 2; ++u) {
        aOff[u] = ldsOffset(&As[u][row * 40 + half * 16]);
        bOff[u] = ldsOffset(&Bs[u][row * 40 + half * 16]);
    }

    // issue one K-tile (32 wide) of A+B into LDS buffer `buf`: 4 async DMAs per wave
    auto issueTile = [&](int k0, int buf) {
        int kk = k0 + half * 16;
        int ky = kk / 2304, kx = (kk / 256) % 9, c0 = kk & 255;
        const unsigned short* srcA =
            h1 + (size_t)((b * 56 + 2 * oy + ky) * 56 + (2 * ox + kx)) * 256 + c0;
        asyncLoadB128(aOff[buf],      srcA);
        asyncLoadB128(aOff[buf] + 16, srcA + 8);
        const unsigned short* srcB = wrow + k0;
        asyncLoadB128(bOff[buf],      srcB);
        asyncLoadB128(bOff[buf] + 16, srcB + 8);
    };

    issueTile(0, 0);   // prologue: tile 0 -> buffer 0

    for (int k0 = 0; k0 < 20736; k0 += 32) {
        const int buf = (k0 >> 5) & 1;
        if (k0 + 32 < 20736) {
            issueTile(k0 + 32, buf ^ 1);  // prefetch next tile into other buffer
            // retire current tile's 4 DMAs; leave the 4 just-issued in flight
            asm volatile("s_wait_asynccnt 0x4" ::: "memory");
        } else {
            asm volatile("s_wait_asynccnt 0x0" ::: "memory");
        }
        __syncthreads();   // all waves' DMAs for `buf` complete -> data visible

        const unsigned short* Ab = As[buf];
        const unsigned short* Bb = Bs[buf];
        v16bf aF[2], bF[4];
        #pragma unroll
        for (int i = 0; i < 2; ++i) {
            int ml = wm * 32 + i * 16 + (lane & 15);
            int kh = lane >> 4;
            Pack32 p;
            p.lo = *(const uint4*)&Ab[ml * 40 + kh * 16];
            p.hi = *(const uint4*)&Ab[ml * 40 + kh * 16 + 8];
            aF[i] = __builtin_bit_cast(v16bf, p);
        }
        #pragma unroll
        for (int j = 0; j < 4; ++j) {
            int nl = wn * 64 + j * 16 + (lane & 15);
            int ks = (lane >> 4) * 16;
            Pack32 p;
            p.lo = *(const uint4*)&Bb[nl * 40 + ks];
            p.hi = *(const uint4*)&Bb[nl * 40 + ks + 8];
            bF[j] = __builtin_bit_cast(v16bf, p);
        }
        #pragma unroll
        for (int i = 0; i < 2; ++i)
            #pragma unroll
            for (int j = 0; j < 4; ++j)
                acc[i][j] = __builtin_amdgcn_wmma_f32_16x16x32_bf16(
                    false, aF[i], false, bF[j], (short)0, acc[i][j], false, false);
        __syncthreads();   // everyone done reading `buf` before it is overwritten
    }

    #pragma unroll
    for (int i = 0; i < 2; ++i)
        #pragma unroll
        for (int j = 0; j < 4; ++j)
            #pragma unroll
            for (int v = 0; v < 8; ++v) {
                int mg = m0 + wm * 32 + i * 16 + v + (lane >> 4) * 8;
                int ng = n0 + wn * 64 + j * 16 + (lane & 15);
                h2[(size_t)mg * 256 + ng] = acc[i][j][v] + bias[ng];
            }
}

// ================= capsule squash (in place on h2 NHWC) =================
__global__ void squash_caps_k(float* __restrict__ h2, int total) {
    int i = blockIdx.x * blockDim.x + threadIdx.x;
    if (i >= total) return;
    int r  = i & 31;
    int bp = i >> 5;                 // b*576 + p
    float* ptr = h2 + (size_t)bp * 256 + r * 8;
    float v[8], nn = 0.f;
    #pragma unroll
    for (int d = 0; d < 8; ++d) { v[d] = ptr[d]; nn += v[d] * v[d]; }
    float sc = (nn / (1.f + nn)) * rsqrtf(nn + 1e-8f);
    #pragma unroll
    for (int d = 0; d < 8; ++d) ptr[d] = sc * v[d];
}

// ================= dynamic routing =================
__global__ void route_partial_k(
    const float* __restrict__ caps,   // h2: [32][576][256]
    const float* __restrict__ rw,     // [32][10][8][16]
    const float* __restrict__ blog,   // [32][18432][10]
    float* __restrict__ partial)      // [32][128][160]
{
    const int chunk = blockIdx.x, r = blockIdx.y, b = blockIdx.z;
    const int tid = threadIdx.x;
    const int c = tid >> 4, o = tid & 15;
    float w[8];
    #pragma unroll
    for (int i = 0; i < 8; ++i) w[i] = rw[((r * 10 + c) * 8 + i) * 16 + o];
    float sacc = 0.f;
    const int p0 = chunk * 144;
    for (int pp = 0; pp < 144; ++pp) {
        int p = p0 + pp;
        const float* cv = caps + ((size_t)(b * 576 + p)) * 256 + r * 8;
        float u = 0.f;
        #pragma unroll
        for (int i = 0; i < 8; ++i) u += cv[i] * w[i];
        int n = r * 576 + p;
        const float* bl = blog + ((size_t)b * 18432 + n) * 10;
        float mx = bl[0];
        #pragma unroll
        for (int q = 1; q < 10; ++q) mx = fmaxf(mx, bl[q]);
        float den = 0.f;
        #pragma unroll
        for (int q = 0; q < 10; ++q) den += __expf(bl[q] - mx);
        sacc += (__expf(bl[c] - mx) / den) * u;
    }
    partial[((size_t)(b * 128) + (r * 4 + chunk)) * 160 + tid] = sacc;
}

__global__ void reduce_squash_k(const float* __restrict__ partial, float* __restrict__ vout) {
    const int b = blockIdx.x;
    const int tid = threadIdx.x;
    float s = 0.f;
    for (int j = 0; j < 128; ++j) s += partial[((size_t)(b * 128) + j) * 160 + tid];
    float n2 = s * s;
    #pragma unroll
    for (int off = 8; off; off >>= 1) n2 += __shfl_xor(n2, off, 16);
    float sc = (n2 / (1.f + n2)) * rsqrtf(n2 + 1e-8f);
    vout[b * 160 + tid] = sc * s;
}

__global__ void b_update_k(
    const float* __restrict__ caps, const float* __restrict__ rw,
    const float* __restrict__ v, float* __restrict__ blog)
{
    const int chunk = blockIdx.x, r = blockIdx.y, b = blockIdx.z;
    const int tid = threadIdx.x;
    const int c = tid >> 4, o = tid & 15;
    float w[8];
    #pragma unroll
    for (int i = 0; i < 8; ++i) w[i] = rw[((r * 10 + c) * 8 + i) * 16 + o];
    const float vv = v[b * 160 + tid];
    const int p0 = chunk * 144;
    for (int pp = 0; pp < 144; ++pp) {
        int p = p0 + pp;
        const float* cv = caps + ((size_t)(b * 576 + p)) * 256 + r * 8;
        float u = 0.f;
        #pragma unroll
        for (int i = 0; i < 8; ++i) u += cv[i] * w[i];
        float d = u * vv;
        #pragma unroll
        for (int off = 8; off; off >>= 1) d += __shfl_xor(d, off, 16);
        if (o == 0) {
            int n = r * 576 + p;
            blog[((size_t)b * 18432 + n) * 10 + c] += d;
        }
    }
}

// ================= host launch =================
extern "C" void kernel_launch(void* const* d_in, const int* in_sizes, int n_in,
                              void* d_out, int out_size, void* d_ws, size_t ws_size,
                              hipStream_t stream) {
    const float* x  = (const float*)d_in[0];
    const float* w1 = (const float*)d_in[1];
    const float* b1 = (const float*)d_in[2];
    const float* w2 = (const float*)d_in[3];
    const float* b2 = (const float*)d_in[4];
    const float* rw = (const float*)d_in[5];
    float* out = (float*)d_out;

    char* ws = (char*)d_ws;
    size_t off = 0;
    auto take = [&](size_t bytes) -> void* {
        void* p = ws + off;
        off = (off + bytes + 255) & ~(size_t)255;
        return p;
    };
    unsigned short* xb   = (unsigned short*)take((size_t)393216 * 2);      // [32,64,64,3] bf16
    unsigned short* w1t  = (unsigned short*)take((size_t)256 * 256 * 2);   // [256][256] bf16
    unsigned short* w2t  = (unsigned short*)take((size_t)256 * 20736 * 2); // [256][20736] bf16
    unsigned short* h1   = (unsigned short*)take((size_t)25690112 * 2);    // [32,56,56,256] bf16
    float* h2    = (float*)take((size_t)4718592 * 4);                      // [32,24,24,256] f32
    float* blog  = (float*)take((size_t)5898240 * 4);                      // [32,18432,10]
    float* part  = (float*)take((size_t)655360 * 4);                       // [32][128][160]
    float* vbuf  = (float*)take((size_t)5120 * 4);                         // [32][10][16]

    // layout conversions
    cvt_x_k <<<(393216 + 255) / 256, 256, 0, stream>>>(x, xb, 393216);
    cvt_w1_k<<<(65536 + 255) / 256, 256, 0, stream>>>(w1, w1t);
    cvt_w2_k<<<(5308416 + 255) / 256, 256, 0, stream>>>(w2, w2t, 5308416);

    // conv1: M=100352 (784 tiles) x N=256 (2 tiles)
    conv1_wmma_k<<<dim3(784, 2), 256, 0, stream>>>(xb, w1t, b1, h1);
    // conv2: M=18432 (144 tiles) x N=256 (2 tiles), K=20736, async double-buffered
    conv2_wmma_k<<<dim3(144, 2), 256, 0, stream>>>(h1, w2t, b2, h2);

    // squash primary capsules in place
    squash_caps_k<<<(589824 + 255) / 256, 256, 0, stream>>>(h2, 589824);

    // routing: b_log = 0, 3 iterations
    zero_f32_k<<<(5898240 + 255) / 256, 256, 0, stream>>>(blog, 5898240);
    for (int it = 0; it < 3; ++it) {
        route_partial_k<<<dim3(4, 32, 32), 160, 0, stream>>>(h2, rw, blog, part);
        float* vo = (it == 2) ? out : vbuf;
        reduce_squash_k<<<32, 160, 0, stream>>>(part, vo);
        if (it < 2)
            b_update_k<<<dim3(4, 32, 32), 160, 0, stream>>>(h2, rw, vbuf, blog);
    }
}
